// TileSelfAttention_43413529428629
// MI455X (gfx1250) — matrix operand order repaired and checked by reading, not compiled, not run on hardware
//
#include <hip/hip_runtime.h>
#include <hip/hip_bf16.h>

// ---------------------------------------------------------------------------
// TileSelfAttention for MI455X (gfx1250, wave32, WMMA).
//
// Pipeline (all on `stream`):
//   0) wsplit:   Wk|Wq|Wv (f32) -> concatenated hi/lo bf16 planes (1536x1024)
//   1) qkv_gemm: Y = x @ Wcat^T + b  via v_wmma_f32_16x16x32_bf16 with
//                hi/lo split-bf16 (3 products) for ~fp32 accuracy.
//                Double-buffered LDS, 2 barriers per K-step, global prefetch.
//                K,Q stored (b,t,c) hi/lo; V stored TRANSPOSED (b,c,t) hi/lo
//                with packed 16B stores.
//   2) kq_logits: aw[b,t,s] = (K[b]·Q[b]^T)/sqrt(512), double-buffered LDS.
//   3) softmax_b: softmax over the BATCH axis (b) per (t,s); writes
//                awT[b,s,t] as hi/lo bf16 (K-contiguous for stage 4).
//   4) av_out:   out[b,c,s] = sum_t Vt[b,c,t] * awT[b,s,t]  (f32 out)
//
// All GEMMs are NT (both operands K-contiguous) which maps 1:1 onto the
// CDNA5 WMMA 16-bit fragment layouts (A: two 16B K-chunks per lane,
// B: two 16B K-chunks per lane).
//
// Workspace use ~224 MB (assumed available; compile-only loop cannot verify).
// ---------------------------------------------------------------------------

typedef __attribute__((ext_vector_type(16))) __bf16 v16bf;
typedef __attribute__((ext_vector_type(8)))  float  v8f;

#define C_IN    1024
#define C_OUT   512
#define NTILES  64
#define BATCH   512
#define MROWS   32768            // BATCH * NTILES
#define NCAT    1536             // 3 * C_OUT
#define LDSS    40               // padded LDS row stride (elements); 80B, 16B-aligned

union Frag {
    v16bf v;
    uint4 q[2];
};

union Pack8 {
    uint4  q;
    __bf16 e[8];
};

__device__ __forceinline__ void split2(float x, __bf16& h, __bf16& l) {
    __bf16 hh = (__bf16)x;
    h = hh;
    l = (__bf16)(x - (float)hh);
}

// A-matrix 16x32 bf16 fragment (M=16, K=32), row-major source, K contiguous.
// lane&15 = row(M), lane>>4 selects K-halves: VGPR0-3 <- K[ch*8 .. +8),
// VGPR4-7 <- K[16+ch*8 .. +8).
__device__ __forceinline__ v16bf load_frag_A(const __bf16* base, int rowStride, int lane) {
    int row = lane & 15;
    int ch  = lane >> 4;
    const __bf16* p = base + (size_t)row * rowStride + ch * 8;
    Frag f;
    f.q[0] = *(const uint4*)(p);
    f.q[1] = *(const uint4*)(p + 16);
    return f.v;
}

// B-matrix 32x16 bf16 fragment (K=32, N=16) for NT GEMM: source stored as
// 16 rows (N) x K contiguous. lane&15 = n, lane>>4 = kh:
// VGPR0-3 <- K[kh*8 .. +8), VGPR4-7 <- K[16+kh*8 .. +8) (matches A).
__device__ __forceinline__ v16bf load_frag_B(const __bf16* base, int rowStride, int lane) {
    int row = lane & 15;
    int kh  = lane >> 4;
    const __bf16* p = base + (size_t)row * rowStride + kh * 8;
    Frag f;
    f.q[0] = *(const uint4*)(p);
    f.q[1] = *(const uint4*)(p + 16);
    return f.v;
}

__device__ __forceinline__ v8f wmma_bf16(v16bf a, v16bf b, v8f c) {
    return __builtin_amdgcn_wmma_f32_16x16x32_bf16(false, a, false, b, (short)0, c, false, false);
}

// ---------------------------------------------------------------------------
// Stage 0: split concatenated weights into hi/lo bf16 planes.
// ---------------------------------------------------------------------------
__global__ __launch_bounds__(256) void wsplit(
    const float* __restrict__ Wk, const float* __restrict__ Wq,
    const float* __restrict__ Wv,
    __bf16* __restrict__ Whi, __bf16* __restrict__ Wlo)
{
    size_t i = (size_t)blockIdx.x * 256 + threadIdx.x;   // [0, 1536*1024)
    const size_t R = (size_t)C_OUT * C_IN;               // 524288
    float v;
    if (i < R)           v = Wk[i];
    else if (i < 2 * R)  v = Wq[i - R];
    else                 v = Wv[i - 2 * R];
    split2(v, Whi[i], Wlo[i]);
}

// ---------------------------------------------------------------------------
// Stage 1: QKV GEMM.  Y[m,n] = sum_k x[m,k] * Wcat[n,k] + bias[n].
// Block tile 128(M) x 64(N), 8 waves (4x2), wave tile 32x32 (2x2 WMMA tiles).
// Double-buffered LDS, two barriers per 32-wide K-step; next tile staged
// into registers while WMMAs consume the current buffer.
// ---------------------------------------------------------------------------
__global__ __launch_bounds__(256) void qkv_gemm(
    const float*  __restrict__ x,
    const __bf16* __restrict__ Whi, const __bf16* __restrict__ Wlo,
    const float*  __restrict__ bk,  const float*  __restrict__ bq,
    const float*  __restrict__ bv,
    __bf16* __restrict__ Khi,  __bf16* __restrict__ Klo,
    __bf16* __restrict__ Qhi,  __bf16* __restrict__ Qlo,
    __bf16* __restrict__ Vthi, __bf16* __restrict__ Vtlo)
{
    __shared__ __bf16 Ah[2][128 * LDSS], Al[2][128 * LDSS];
    __shared__ __bf16 Bh[2][64 * LDSS],  Bl[2][64 * LDSS];

    const int tid  = threadIdx.x;
    const int lane = tid & 31;
    const int wave = tid >> 5;
    const int wm   = wave >> 1;        // 0..3
    const int wn   = wave & 1;         // 0..1
    const int Mbase = blockIdx.x * 128;
    const int Nbase = blockIdx.y * 64; // within [0, 1536)

    // cooperative staging geometry
    const int rA = tid >> 3;           // A: 4 rows x float4 per thread
    const int cA = (tid & 7) * 4;
    const int rB = tid >> 2;           // B: 1 row x uint4 per thread (per plane)
    const int cB = (tid & 3) * 8;

    v8f acc[2][2] = {};

    float4 ax[4];
    uint4  wh, wl;

    // prologue: stage K-step 0
    #pragma unroll
    for (int i = 0; i < 4; ++i)
        ax[i] = *(const float4*)(x + (size_t)(Mbase + rA + i * 32) * C_IN + cA);
    wh = *(const uint4*)(Whi + (size_t)(Nbase + rB) * C_IN + cB);
    wl = *(const uint4*)(Wlo + (size_t)(Nbase + rB) * C_IN + cB);

    const int KSTEPS = C_IN / 32;      // 32
    for (int k = 0; k < KSTEPS; ++k) {
        const int buf = k & 1;

        // --- commit staged registers into LDS[buf] ---
        #pragma unroll
        for (int i = 0; i < 4; ++i) {
            const int o = (rA + i * 32) * LDSS + cA;
            split2(ax[i].x, Ah[buf][o + 0], Al[buf][o + 0]);
            split2(ax[i].y, Ah[buf][o + 1], Al[buf][o + 1]);
            split2(ax[i].z, Ah[buf][o + 2], Al[buf][o + 2]);
            split2(ax[i].w, Ah[buf][o + 3], Al[buf][o + 3]);
        }
        *(uint4*)&Bh[buf][rB * LDSS + cB] = wh;
        *(uint4*)&Bl[buf][rB * LDSS + cB] = wl;
        __syncthreads();

        // --- stage next K-step into registers (latency hidden by WMMAs) ---
        if (k + 1 < KSTEPS) {
            const int k0 = (k + 1) * 32;
            #pragma unroll
            for (int i = 0; i < 4; ++i)
                ax[i] = *(const float4*)(x + (size_t)(Mbase + rA + i * 32) * C_IN + k0 + cA);
            wh = *(const uint4*)(Whi + (size_t)(Nbase + rB) * C_IN + k0 + cB);
            wl = *(const uint4*)(Wlo + (size_t)(Nbase + rB) * C_IN + k0 + cB);
            if (k + 3 < KSTEPS) {
                // keep the x stream ahead in L2 (global_prefetch_b8)
                __builtin_prefetch(x + (size_t)(Mbase + rA) * C_IN + (k + 3) * 32 + cA, 0, 3);
            }
        }

        // --- WMMA on LDS[buf] ---
        v16bf fah[2], fal[2], fbh[2], fbl[2];
        #pragma unroll
        for (int i = 0; i < 2; ++i) {
            const int mo = (wm * 32 + i * 16) * LDSS;
            fah[i] = load_frag_A(&Ah[buf][mo], LDSS, lane);
            fal[i] = load_frag_A(&Al[buf][mo], LDSS, lane);
        }
        #pragma unroll
        for (int j = 0; j < 2; ++j) {
            const int no = (wn * 32 + j * 16) * LDSS;
            fbh[j] = load_frag_B(&Bh[buf][no], LDSS, lane);
            fbl[j] = load_frag_B(&Bl[buf][no], LDSS, lane);
        }
        #pragma unroll
        for (int i = 0; i < 2; ++i)
            #pragma unroll
            for (int j = 0; j < 2; ++j) {
                acc[i][j] = wmma_bf16(fah[i], fbh[j], acc[i][j]);
                acc[i][j] = wmma_bf16(fah[i], fbl[j], acc[i][j]);
                acc[i][j] = wmma_bf16(fal[i], fbh[j], acc[i][j]);
            }
        __syncthreads();   // protects LDS[buf^1] (written next iter) vs this iter's reads two back
    }

    // --- bias + hi/lo store (K/Q direct, V transposed to (b,c,t)) ---
    const int widx = Nbase >> 9;                     // 0:K 1:Q 2:V
    const float* bias = (widx == 0) ? bk : ((widx == 1) ? bq : bv);

    const int half = lane >> 4;
    const int col  = lane & 15;

    if (widx == 2) {
        // V: packed 16B stores, 8 contiguous t's at fixed c
        #pragma unroll
        for (int j = 0; j < 2; ++j) {
            const int c = (Nbase + wn * 32 + j * 16 + col) & (C_OUT - 1);
            const float bconst = bias[c];
            #pragma unroll
            for (int i = 0; i < 2; ++i) {
                const int mrun = Mbase + wm * 32 + i * 16 + half * 8; // multiple of 8
                Pack8 ph, pl;
                #pragma unroll
                for (int r = 0; r < 8; ++r)
                    split2(acc[i][j][r] + bconst, ph.e[r], pl.e[r]);
                const size_t idx = (size_t)(mrun >> 6) * (C_OUT * NTILES)
                                 + (size_t)c * NTILES + (mrun & 63);
                *(uint4*)(Vthi + idx) = ph.q;
                *(uint4*)(Vtlo + idx) = pl.q;
            }
        }
    } else {
        __bf16* dsthi = (widx == 0) ? Khi : Qhi;
        __bf16* dstlo = (widx == 0) ? Klo : Qlo;
        #pragma unroll
        for (int j = 0; j < 2; ++j) {
            const int c = (Nbase + wn * 32 + j * 16 + col) & (C_OUT - 1);
            const float bconst = bias[c];
            #pragma unroll
            for (int i = 0; i < 2; ++i) {
                #pragma unroll
                for (int r = 0; r < 8; ++r) {
                    const int m = Mbase + wm * 32 + i * 16 + half * 8 + r;
                    __bf16 h, l;
                    split2(acc[i][j][r] + bconst, h, l);
                    const size_t idx = (size_t)m * C_OUT + c;        // (b,t,c)
                    dsthi[idx] = h;
                    dstlo[idx] = l;
                }
            }
        }
    }
}

// ---------------------------------------------------------------------------
// Stage 2: logits aw[b,t,s] = (sum_c K[b,t,c]*Q[b,s,c]) / sqrt(C_OUT)
// One block per batch; 8 waves; wave -> (tm, 2 tn tiles) of the 4x4 tile grid.
// Double-buffered LDS, two barriers per 32-wide c-step.
// ---------------------------------------------------------------------------
__global__ __launch_bounds__(256) void kq_logits(
    const __bf16* __restrict__ Khi, const __bf16* __restrict__ Klo,
    const __bf16* __restrict__ Qhi, const __bf16* __restrict__ Qlo,
    float* __restrict__ aw)
{
    __shared__ __bf16 Kh[2][64 * LDSS], Kl[2][64 * LDSS];
    __shared__ __bf16 Qh[2][64 * LDSS], Ql[2][64 * LDSS];

    const int b    = blockIdx.x;
    const int tid  = threadIdx.x;
    const int lane = tid & 31;
    const int wave = tid >> 5;
    const int tm   = wave & 3;
    const int tn0  = (wave >> 2) * 2;
    const size_t bbase = (size_t)b * NTILES * C_OUT;

    const int r  = tid >> 2;
    const int c8 = (tid & 3) * 8;

    v8f acc[2] = {};

    uint4 skh, skl, sqh, sql;
    {
        const size_t g = bbase + (size_t)r * C_OUT + c8;
        skh = *(const uint4*)(Khi + g);
        skl = *(const uint4*)(Klo + g);
        sqh = *(const uint4*)(Qhi + g);
        sql = *(const uint4*)(Qlo + g);
    }

    const int KSTEPS = C_OUT / 32;     // 16
    for (int k = 0; k < KSTEPS; ++k) {
        const int buf = k & 1;
        const int o = r * LDSS + c8;
        *(uint4*)&Kh[buf][o] = skh;
        *(uint4*)&Kl[buf][o] = skl;
        *(uint4*)&Qh[buf][o] = sqh;
        *(uint4*)&Ql[buf][o] = sql;
        __syncthreads();

        if (k + 1 < KSTEPS) {
            const size_t g = bbase + (size_t)r * C_OUT + (k + 1) * 32 + c8;
            skh = *(const uint4*)(Khi + g);
            skl = *(const uint4*)(Klo + g);
            sqh = *(const uint4*)(Qhi + g);
            sql = *(const uint4*)(Qlo + g);
        }

        const v16bf ah = load_frag_A(&Kh[buf][tm * 16 * LDSS], LDSS, lane);
        const v16bf al = load_frag_A(&Kl[buf][tm * 16 * LDSS], LDSS, lane);
        #pragma unroll
        for (int j = 0; j < 2; ++j) {
            const int no = (tn0 + j) * 16 * LDSS;
            const v16bf bh = load_frag_B(&Qh[buf][no], LDSS, lane);
            const v16bf bl = load_frag_B(&Ql[buf][no], LDSS, lane);
            acc[j] = wmma_bf16(ah, bh, acc[j]);
            acc[j] = wmma_bf16(ah, bl, acc[j]);
            acc[j] = wmma_bf16(al, bh, acc[j]);
        }
        __syncthreads();
    }

    const float scale = 0.044194173824159216f;   // 1/sqrt(512)
    const int half = lane >> 4;
    const int col  = lane & 15;
    #pragma unroll
    for (int j = 0; j < 2; ++j) {
        const int s = (tn0 + j) * 16 + col;
        #pragma unroll
        for (int rr = 0; rr < 8; ++rr) {
            const int t = tm * 16 + half * 8 + rr;
            aw[(size_t)b * (NTILES * NTILES) + t * NTILES + s] = acc[j][rr] * scale;
        }
    }
}

// ---------------------------------------------------------------------------
// Stage 3: softmax over the BATCH axis (b) per (t,s).  Writes awT[b,s,t]
// as hi/lo bf16 (t contiguous -> B-fragment-friendly for stage 4).
// Grid: 64 blocks (t); block (64 s-lanes, 8 b-rows).
// ---------------------------------------------------------------------------
__global__ __launch_bounds__(512) void softmax_b(
    const float* __restrict__ aw,
    __bf16* __restrict__ awThi, __bf16* __restrict__ awTlo)
{
    __shared__ float red[8][64];
    const int t  = blockIdx.x;
    const int s  = threadIdx.x;
    const int br = threadIdx.y;
    const size_t col = (size_t)t * NTILES + s;

    float m = -1e30f;
    for (int b = br; b < BATCH; b += 8)
        m = fmaxf(m, aw[(size_t)b * (NTILES * NTILES) + col]);
    red[br][s] = m;
    __syncthreads();
    if (br == 0) {
        float mm = red[0][s];
        #pragma unroll
        for (int i = 1; i < 8; ++i) mm = fmaxf(mm, red[i][s]);
        red[0][s] = mm;
    }
    __syncthreads();
    const float mx = red[0][s];
    __syncthreads();

    float sum = 0.f;
    for (int b = br; b < BATCH; b += 8)
        sum += __expf(aw[(size_t)b * (NTILES * NTILES) + col] - mx);
    red[br][s] = sum;
    __syncthreads();
    if (br == 0) {
        float ss = red[0][s];
        #pragma unroll
        for (int i = 1; i < 8; ++i) ss += red[i][s];
        red[0][s] = ss;
    }
    __syncthreads();
    const float inv = 1.f / red[0][s];

    for (int b = br; b < BATCH; b += 8) {
        const float w = __expf(aw[(size_t)b * (NTILES * NTILES) + col] - mx) * inv;
        const size_t o = (size_t)b * (NTILES * NTILES) + (size_t)s * NTILES + t;
        split2(w, awThi[o], awTlo[o]);
    }
}

// ---------------------------------------------------------------------------
// Stage 4: out[b,c,s] = sum_t Vt[b,c,t] * awT[b,s,t]  (K = 64, 2 WMMA steps)
// Grid (4, 512): (c-block of 128, batch). 8 waves; wave = one 16-row c-tile
// across all 4 s-tiles. Fragments loaded straight from global (K-contiguous).
// ---------------------------------------------------------------------------
__global__ __launch_bounds__(256) void av_out(
    const __bf16* __restrict__ Vthi, const __bf16* __restrict__ Vtlo,
    const __bf16* __restrict__ awThi, const __bf16* __restrict__ awTlo,
    float* __restrict__ out)
{
    const int b    = blockIdx.y;
    const int tid  = threadIdx.x;
    const int lane = tid & 31;
    const int wave = tid >> 5;
    const int c0   = blockIdx.x * 128 + wave * 16;

    const __bf16* Vh  = Vthi  + (size_t)b * (C_OUT * NTILES);
    const __bf16* Vl  = Vtlo  + (size_t)b * (C_OUT * NTILES);
    const __bf16* Awh = awThi + (size_t)b * (NTILES * NTILES);
    const __bf16* Awl = awTlo + (size_t)b * (NTILES * NTILES);

    v8f acc[4] = {};

    #pragma unroll
    for (int t0 = 0; t0 < NTILES; t0 += 32) {
        const v16bf ah = load_frag_A(Vh + (size_t)c0 * NTILES + t0, NTILES, lane);
        const v16bf al = load_frag_A(Vl + (size_t)c0 * NTILES + t0, NTILES, lane);
        #pragma unroll
        for (int j = 0; j < 4; ++j) {
            const v16bf bh = load_frag_B(Awh + (size_t)j * 16 * NTILES + t0, NTILES, lane);
            const v16bf bl = load_frag_B(Awl + (size_t)j * 16 * NTILES + t0, NTILES, lane);
            acc[j] = wmma_bf16(ah, bh, acc[j]);
            acc[j] = wmma_bf16(ah, bl, acc[j]);
            acc[j] = wmma_bf16(al, bh, acc[j]);
        }
    }

    const int half = lane >> 4;
    const int col  = lane & 15;
    #pragma unroll
    for (int j = 0; j < 4; ++j) {
        const int s = j * 16 + col;
        #pragma unroll
        for (int r = 0; r < 8; ++r) {
            const int c = c0 + half * 8 + r;
            out[(size_t)b * (C_OUT * NTILES) + (size_t)c * NTILES + s] = acc[j][r];
        }
    }
}

// ---------------------------------------------------------------------------
extern "C" void kernel_launch(void* const* d_in, const int* in_sizes, int n_in,
                              void* d_out, int out_size, void* d_ws, size_t ws_size,
                              hipStream_t stream) {
    const float* x  = (const float*)d_in[0];
    const float* Wk = (const float*)d_in[1];
    const float* bk = (const float*)d_in[2];
    const float* Wq = (const float*)d_in[3];
    const float* bq = (const float*)d_in[4];
    const float* Wv = (const float*)d_in[5];
    const float* bv = (const float*)d_in[6];
    float* out = (float*)d_out;

    char* ws = (char*)d_ws;
    auto carve = [&](size_t bytes) -> char* {
        char* p = ws;
        ws += (bytes + 255) & ~(size_t)255;
        return p;
    };
    const size_t kq_elems = (size_t)MROWS * C_OUT;      // 16.78M
    __bf16* Whi  = (__bf16*)carve((size_t)NCAT * C_IN * 2);
    __bf16* Wlo  = (__bf16*)carve((size_t)NCAT * C_IN * 2);
    __bf16* Khi  = (__bf16*)carve(kq_elems * 2);
    __bf16* Klo  = (__bf16*)carve(kq_elems * 2);
    __bf16* Qhi  = (__bf16*)carve(kq_elems * 2);
    __bf16* Qlo  = (__bf16*)carve(kq_elems * 2);
    __bf16* Vthi = (__bf16*)carve(kq_elems * 2);
    __bf16* Vtlo = (__bf16*)carve(kq_elems * 2);
    float*  aw   = (float*) carve((size_t)BATCH * NTILES * NTILES * 4);
    __bf16* awThi= (__bf16*)carve((size_t)BATCH * NTILES * NTILES * 2);
    __bf16* awTlo= (__bf16*)carve((size_t)BATCH * NTILES * NTILES * 2);
    (void)ws_size; (void)in_sizes; (void)n_in; (void)out_size;

    // 0) weight split: 1536*1024 / 256 = 6144 blocks
    wsplit<<<6144, 256, 0, stream>>>(Wk, Wq, Wv, Whi, Wlo);

    // 1) QKV projections: grid (32768/128, 1536/64)
    qkv_gemm<<<dim3(MROWS / 128, NCAT / 64), 256, 0, stream>>>(
        x, Whi, Wlo, bk, bq, bv, Khi, Klo, Qhi, Qlo, Vthi, Vtlo);

    // 2) logits: one block per batch
    kq_logits<<<BATCH, 256, 0, stream>>>(Khi, Klo, Qhi, Qlo, aw);

    // 3) softmax over batch axis
    softmax_b<<<NTILES, dim3(64, 8), 0, stream>>>(aw, awThi, awTlo);

    // 4) V^T @ aw
    av_out<<<dim3(4, BATCH), 256, 0, stream>>>(Vthi, Vtlo, awThi, awTlo, out);
}
